// LightGCN_15642270892369
// MI455X (gfx1250) — matrix operand order repaired and checked by reading, not compiled, not run on hardware
//
#include <hip/hip_runtime.h>
#include <hip/hip_bf16.h>

#define NUSERS  100000
#define NITEMS  50000
#define DIM     64
#define NNODES  (NUSERS + NITEMS)
#define NEDGES  6400000
#define QB      1024
#define NLAYERS 3

#define MTILES (QB / 16)            // 64
#define NTILES ((NITEMS + 15) / 16) // 3125
#define WPB    8                    // waves per block in GEMM

typedef __attribute__((ext_vector_type(16))) _Float16 v16h;
typedef __attribute__((ext_vector_type(8)))  _Float16 v8h;
typedef __attribute__((ext_vector_type(8)))  float    v8f;

// ---------------------------------------------------------------------------
// 1) acc = x0 = concat(user_emb, item_emb)   (float4 granularity)
// ---------------------------------------------------------------------------
__global__ void init_concat_kernel(const float* __restrict__ ue,
                                   const float* __restrict__ ie,
                                   float* __restrict__ acc,
                                   float* __restrict__ x,
                                   int nq) {
    int q = blockIdx.x * blockDim.x + threadIdx.x;
    if (q >= nq) return;
    float4 v;
    if (q < NUSERS * (DIM / 4)) v = ((const float4*)ue)[q];
    else                        v = ((const float4*)ie)[q - NUSERS * (DIM / 4)];
    ((float4*)acc)[q] = v;
    ((float4*)x)[q]   = v;
}

// ---------------------------------------------------------------------------
// 2) zero the scatter target
// ---------------------------------------------------------------------------
__global__ void zero_kernel(float* __restrict__ p, int nq) {
    int q = blockIdx.x * blockDim.x + threadIdx.x;
    if (q >= nq) return;
    ((float4*)p)[q] = make_float4(0.f, 0.f, 0.f, 0.f);
}

// ---------------------------------------------------------------------------
// 3) SpMM scatter: one wave (32 lanes) per edge, float2 per lane (64 dims).
//    Gather x[src] coalesced (256B/wave), scale by edge weight, scatter-add
//    to x_next[dst] with native f32 global atomics (resolve in 192MB L2;
//    x_next is 38.4MB so the whole scatter target is L2-resident).
// ---------------------------------------------------------------------------
__global__ void spmm_kernel(const float* __restrict__ x,
                            const float* __restrict__ ev,
                            const int*   __restrict__ src,
                            const int*   __restrict__ dst,
                            float* __restrict__ xn,
                            int nEdges) {
    int tid  = blockIdx.x * blockDim.x + threadIdx.x;
    int e    = tid >> 5;
    if (e >= nEdges) return;
    int lane = tid & 31;
    int s = src[e];
    int d = dst[e];
    float w = ev[e];
    const float2 v = *(const float2*)(x + (size_t)s * DIM + 2 * lane);
    float* o = xn + (size_t)d * DIM + 2 * lane;
    unsafeAtomicAdd(o,     v.x * w);   // global_atomic_add_f32
    unsafeAtomicAdd(o + 1, v.y * w);
}

// ---------------------------------------------------------------------------
// 4) acc += x_next
// ---------------------------------------------------------------------------
__global__ void accum_kernel(float* __restrict__ acc,
                             const float* __restrict__ xn, int nq) {
    int q = blockIdx.x * blockDim.x + threadIdx.x;
    if (q >= nq) return;
    float4 a = ((const float4*)acc)[q];
    float4 b = ((const float4*)xn)[q];
    a.x += b.x; a.y += b.y; a.z += b.z; a.w += b.w;
    ((float4*)acc)[q] = a;
}

// ---------------------------------------------------------------------------
// 5) item block of light_out -> f16 (fold the /(NLAYERS+1) scale)
// ---------------------------------------------------------------------------
__global__ void items_f16_kernel(const float* __restrict__ acc,
                                 _Float16* __restrict__ itemsF16) {
    int i = blockIdx.x * blockDim.x + threadIdx.x;
    if (i >= NITEMS * DIM) return;
    itemsF16[i] = (_Float16)(acc[(size_t)NUSERS * DIM + i] * 0.25f);
}

// ---------------------------------------------------------------------------
// 6) gather queried user rows -> f16 (fold scale)
// ---------------------------------------------------------------------------
__global__ void users_f16_kernel(const float* __restrict__ acc,
                                 const int* __restrict__ users,
                                 _Float16* __restrict__ usersF16) {
    int i = blockIdx.x * blockDim.x + threadIdx.x;
    if (i >= QB * DIM) return;
    int b = i >> 6;
    int k = i & (DIM - 1);
    int u = users[b];
    usersF16[i] = (_Float16)(acc[(size_t)u * DIM + k] * 0.25f);
}

// ---------------------------------------------------------------------------
// 7) rating = sigmoid(U @ I^T) via v_wmma_f32_16x16x32_f16.
//    One wave per 16x16 output tile, K=64 -> 2 WMMAs.
//    Fragments: 16-bit A 16x32, B 32x16 (contiguous K per lane half in both), f32 C/D.
//    Epilogue uses v_exp_f32 + v_rcp_f32 (fast sigmoid) so the stage stays
//    bound by the 205MB rating store, not by VALU.
// ---------------------------------------------------------------------------
__global__ void gemm_sigmoid_kernel(const _Float16* __restrict__ U,
                                    const _Float16* __restrict__ I,
                                    float* __restrict__ out) {
    const int wave  = threadIdx.x >> 5;               // 0..7
    const int lane  = threadIdx.x & 31;
    const int mtile = blockIdx.y;
    const int ntile = blockIdx.x * WPB + wave;
    if (ntile >= NTILES) return;                      // wave-uniform: EXEC stays all-1s

    const int hs = lane >> 4;                         // half-wave select
    const int li = lane & 15;

    // A: row M = mtile*16 + li ; lane holds contiguous K[hs*16 .. hs*16+16)
    const _Float16* urow = U + (size_t)(mtile * 16 + li) * DIM;
    // B: col N = ntile*16 + li ; lane holds contiguous K[hs*16 .. hs*16+16)
    const _Float16* irow = I + (size_t)(ntile * 16 + li) * DIM;

    __builtin_prefetch(irow + 16 * DIM, 0, 3);        // global_prefetch_b8 next tile row

    v8f c = {};
#pragma unroll
    for (int t = 0; t < 2; ++t) {                     // K blocks of 32
        union { v16h v; v8h h[2]; } a;
        a.h[0] = *(const v8h*)(urow + t * 32 + hs * 16);
        a.h[1] = *(const v8h*)(urow + t * 32 + hs * 16 + 8);
        v16h b = *(const v16h*)(irow + t * 32 + hs * 16);
        c = __builtin_amdgcn_wmma_f32_16x16x32_f16(
                /*neg_a=*/false, a.v, /*neg_b=*/false, b,
                /*c_mod=*/(short)0, c, /*reuse_a=*/false, /*reuse_b=*/false);
    }

    // D layout: VGPR r, lanes 0-15 -> M=r, lanes 16-31 -> M=r+8
    float* obase = out + (size_t)(mtile * 16 + hs * 8) * NITEMS + ntile * 16 + li;
#pragma unroll
    for (int r = 0; r < 8; ++r) {
        float v = c[r];
        // fast sigmoid: single v_exp_f32 + v_add + v_rcp_f32 (no IEEE div chain)
        v = __builtin_amdgcn_rcpf(1.0f + __expf(-v));
        obase[(size_t)r * NITEMS] = v;
    }
}

// ---------------------------------------------------------------------------
// host-side orchestration (all launches on `stream`; graph-capture safe)
// ---------------------------------------------------------------------------
extern "C" void kernel_launch(void* const* d_in, const int* in_sizes, int n_in,
                              void* d_out, int out_size, void* d_ws, size_t ws_size,
                              hipStream_t stream) {
    const float* user_emb = (const float*)d_in[0];
    const float* item_emb = (const float*)d_in[1];
    const float* edge_val = (const float*)d_in[2];
    const int*   edge_src = (const int*)d_in[3];
    const int*   edge_dst = (const int*)d_in[4];
    const int*   users    = (const int*)d_in[5];
    float*       out      = (float*)d_out;

    const int nEdges = in_sizes[2];                   // 6,400,000

    // workspace layout (all offsets 256B-aligned by construction)
    char* ws = (char*)d_ws;
    const size_t embBytes = (size_t)NNODES * DIM * sizeof(float);   // 38.4 MB
    float*    acc   = (float*)(ws);
    float*    xA    = (float*)(ws + embBytes);
    float*    xB    = (float*)(ws + 2 * embBytes);
    _Float16* uf16  = (_Float16*)(ws + 3 * embBytes);
    _Float16* if16  = (_Float16*)(ws + 3 * embBytes + (size_t)QB * DIM * sizeof(_Float16));

    const int nq = NNODES * (DIM / 4);                // float4 elements: 2.4M
    const int qBlocks = (nq + 255) / 256;

    init_concat_kernel<<<qBlocks, 256, 0, stream>>>(user_emb, item_emb, acc, xA, nq);

    float* xc = xA;
    float* xn = xB;
    for (int l = 0; l < NLAYERS; ++l) {
        zero_kernel<<<qBlocks, 256, 0, stream>>>(xn, nq);
        long long spmmThreads = (long long)nEdges * 32;
        int spmmBlocks = (int)((spmmThreads + 255) / 256);
        spmm_kernel<<<spmmBlocks, 256, 0, stream>>>(xc, edge_val, edge_src, edge_dst, xn, nEdges);
        accum_kernel<<<qBlocks, 256, 0, stream>>>(acc, xn, nq);
        float* t = xc; xc = xn; xn = t;
    }

    items_f16_kernel<<<(NITEMS * DIM + 255) / 256, 256, 0, stream>>>(acc, if16);
    users_f16_kernel<<<(QB * DIM + 255) / 256, 256, 0, stream>>>(acc, users, uf16);

    dim3 g((NTILES + WPB - 1) / WPB, MTILES);         // (391, 64)
    gemm_sigmoid_kernel<<<g, 256, 0, stream>>>(uf16, if16, out);
}
